// LongShortTermMemoryModel_68006512165219
// MI455X (gfx1250) — matrix operand order repaired and checked by reading, not compiled, not run on hardware
//
#include <hip/hip_runtime.h>
#include <hip/hip_bf16.h>

typedef __bf16 bf16;
typedef __attribute__((ext_vector_type(16))) bf16  v16bf;
typedef __attribute__((ext_vector_type(8)))  bf16  v8bf;
typedef __attribute__((ext_vector_type(4)))  bf16  v4bf;
typedef __attribute__((ext_vector_type(8)))  float v8f;
typedef __attribute__((ext_vector_type(4)))  float v4f;

union V16 { v16bf v; v8bf h[2]; };
union V8F { v8f v; v4f h[2]; float f[8]; };

#define T_SEQ 512
#define NBATCH 256
#define INSZ 256
#define HID 128
#define G4H 512
#define OUTSZ 1000
#define WHH_LD 136   // padded LDS row stride (bf16 elems) -> conflict-free b128 tile loads

__device__ __forceinline__ float sigmoidf_(float x) { return 1.0f / (1.0f + __expf(-x)); }
__device__ __forceinline__ float tanhf_(float x) {
  float e = __expf(-2.0f * x);
  return (1.0f - e) / (1.0f + e);
}

// ---------------- Kernel A: one-time conversions -------------------------
__global__ void k_convert(const float* __restrict__ wih, const float* __restrict__ whh,
                          const float* __restrict__ bih, const float* __restrict__ bhh,
                          bf16* __restrict__ wihb, bf16* __restrict__ whhb,
                          float* __restrict__ biasc) {
  int i = blockIdx.x * blockDim.x + threadIdx.x;
  if (i < G4H * INSZ) wihb[i] = (bf16)wih[i];
  if (i < G4H * HID)  whhb[i] = (bf16)whh[i];
  if (i < G4H)        biasc[i] = bih[i] + bhh[i];
}

// ---------------- Kernel B: x_proj = x @ w_ih^T + bias (WMMA bf16) -------
// Grid: 2048 blocks x 512 threads. Block covers 64 rows x all 512 gate cols.
// Output stored in WMMA C-fragment tile order: [mt*32+gt][lane][8 f32].
__global__ __launch_bounds__(512) void k_xproj(const float* __restrict__ x,
                                               const bf16* __restrict__ wihb,
                                               const float* __restrict__ biasc,
                                               float* __restrict__ xproj) {
  const int lane = threadIdx.x & 31;
  const int wave = threadIdx.x >> 5;
  const int mt   = blockIdx.x * 4 + (wave & 3);   // 16-row tile index (0..8191)
  const int gq   = wave >> 2;                     // 0..3 -> gate tiles gq*8 .. gq*8+7
  const int hl   = lane >> 4;                     // lane half
  const int ln   = lane & 15;

  const float* aBase = x + (size_t)(mt * 16 + ln) * INSZ + hl * 8;

  v8f acc[8];
#pragma unroll
  for (int n = 0; n < 8; ++n) {
    float bv = biasc[gq * 128 + n * 16 + ln];     // per-column bias
#pragma unroll
    for (int e = 0; e < 8; ++e) acc[n][e] = bv;
  }

#pragma unroll
  for (int kk = 0; kk < 8; ++kk) {                // K = 256 in steps of 32
    // --- issue ALL loads for this K-step first (one clause, one wait) ---
    const float* ap = aBase + kk * 32;
    v4f a0 = *(const v4f*)(ap);
    v4f a1 = *(const v4f*)(ap + 4);
    v4f a2 = *(const v4f*)(ap + 16);
    v4f a3 = *(const v4f*)(ap + 20);

    V16 Bt[8];                                    // distinct regs per tile
#pragma unroll
    for (int n = 0; n < 8; ++n) {
      const bf16* bp = wihb + (size_t)(gq * 128 + n * 16 + ln) * INSZ + kk * 32 + hl * 8;
      Bt[n].h[0] = *(const v8bf*)bp;              // K = koff .. koff+7
      Bt[n].h[1] = *(const v8bf*)(bp + 16);       // K = koff+16 .. koff+23
    }

    v16bf A;
#pragma unroll
    for (int e = 0; e < 4; ++e) {                 // f32 -> bf16 on the fly
      A[e]      = (bf16)a0[e];
      A[e + 4]  = (bf16)a1[e];
      A[e + 8]  = (bf16)a2[e];
      A[e + 12] = (bf16)a3[e];
    }

#pragma unroll
    for (int n = 0; n < 8; ++n)
      acc[n] = __builtin_amdgcn_wmma_f32_16x16x32_bf16(
          false, A, false, Bt[n].v, (short)0, acc[n], false, false);
  }

#pragma unroll
  for (int n = 0; n < 8; ++n) {
    float* op = xproj + ((size_t)(mt * 32 + gq * 8 + n)) * 256 + lane * 8;
    V8F c; c.v = acc[n];
    *(v4f*)op       = c.h[0];
    *(v4f*)(op + 4) = c.h[1];
  }
}

// ---------------- Kernel C: LSTM recurrence ------------------------------
// 16 independent workgroups (one per 16-row batch tile), 256 threads each.
// Wave jt owns hidden columns [jt*16, jt*16+16). h in LDS (bf16, padded),
// w_hh fragments pinned in REGISTERS for all 512 steps, c in registers,
// x_proj fragments double-buffered. 16 WMMA per wave per timestep.
__global__ __launch_bounds__(256) void k_lstm(const float* __restrict__ xproj,
                                              const bf16* __restrict__ whhb,
                                              float* __restrict__ hlast) {
  __shared__ bf16 sH[16 * WHH_LD];                  // ~4.4 KB

  const int tid  = threadIdx.x;
  const int lane = tid & 31;
  const int jt   = tid >> 5;                        // wave id = hidden col tile 0..7
  const int hl   = lane >> 4;
  const int ln   = lane & 15;
  const int bt   = blockIdx.x;                      // batch tile 0..15

  const int gts[4] = {jt, jt + 8, jt + 16, jt + 24};  // i, f, g, o col tiles

  // w_hh B-fragments: loaded once from global, live in registers 512 steps.
  V16 Bw[4][4];                                     // [gate q][k-step]
#pragma unroll
  for (int q = 0; q < 4; ++q)
#pragma unroll
    for (int kk = 0; kk < 4; ++kk) {
      const bf16* bp = whhb + (size_t)(gts[q] * 16 + ln) * HID + kk * 32 + hl * 8;
      Bw[q][kk].h[0] = *(const v8bf*)bp;
      Bw[q][kk].h[1] = *(const v8bf*)(bp + 16);
    }

  for (int i = tid; i < 16 * WHH_LD; i += 256) sH[i] = (bf16)0.0f;  // h0 = 0
  __syncthreads();

  v8f c;
#pragma unroll
  for (int e = 0; e < 8; ++e) c[e] = 0.0f;          // c0 = 0

  // Preload x_proj fragments for t = 0 (C-fragment layout: 32 B per lane).
  V8F xp[4], xpn[4];
#pragma unroll
  for (int q = 0; q < 4; ++q) {
    const float* p = xproj + ((size_t)(bt * 32 + gts[q])) * 256 + lane * 8;
    xp[q].h[0] = *(const v4f*)p;
    xp[q].h[1] = *(const v4f*)(p + 4);
  }

  for (int t = 0; t < T_SEQ; ++t) {
    v8f acc4[4];
#pragma unroll
    for (int q = 0; q < 4; ++q)
#pragma unroll
      for (int e = 0; e < 8; ++e) acc4[q][e] = 0.0f;

#pragma unroll
    for (int kk = 0; kk < 4; ++kk) {                // K = 128 in steps of 32
      const bf16* apA = sH + ln * WHH_LD + kk * 32 + hl * 8;
      V16 A;
      A.h[0] = *(const v8bf*)apA;                   // K = kb .. kb+7
      A.h[1] = *(const v8bf*)(apA + 16);            // K = kb+16 .. kb+23
#pragma unroll
      for (int q = 0; q < 4; ++q)
        acc4[q] = __builtin_amdgcn_wmma_f32_16x16x32_bf16(
            false, A.v, false, Bw[q][kk].v, (short)0, acc4[q], false, false);
    }

    // Prefetch next step's x_proj fragments (hidden behind this whole step).
    if (t + 1 < T_SEQ) {
      const int mtn = (t + 1) * 16 + bt;
#pragma unroll
      for (int q = 0; q < 4; ++q) {
        const float* p = xproj + ((size_t)(mtn * 32 + gts[q])) * 256 + lane * 8;
        xpn[q].h[0] = *(const v4f*)p;
        xpn[q].h[1] = *(const v4f*)(p + 4);
      }
    }
    __syncthreads();                                // all sH reads complete

#pragma unroll
    for (int e = 0; e < 8; ++e) {
      float gi = sigmoidf_(acc4[0][e] + xp[0].f[e]);
      float gf = sigmoidf_(acc4[1][e] + xp[1].f[e]);
      float gg = tanhf_  (acc4[2][e] + xp[2].f[e]);
      float go = sigmoidf_(acc4[3][e] + xp[3].f[e]);
      float cc = gf * c[e] + gi * gg;
      c[e] = cc;
      float h = go * tanhf_(cc);
      int row = e + hl * 8;                         // C-fragment row mapping
      int col = jt * 16 + ln;
      sH[row * WHH_LD + col] = (bf16)h;
      if (t == T_SEQ - 1) hlast[(bt * 16 + row) * HID + col] = h;
    }
    __syncthreads();                                // sH writes visible

#pragma unroll
    for (int q = 0; q < 4; ++q) xp[q] = xpn[q];
  }
}

// ---------------- Kernel D: dense + softmax ------------------------------
__global__ __launch_bounds__(256) void k_dense(const float* __restrict__ hlast,
                                               const float* __restrict__ wd,
                                               const float* __restrict__ bd,
                                               float* __restrict__ out) {
  __shared__ float sh[HID];
  __shared__ float red[256];
  const int b = blockIdx.x, tid = threadIdx.x;
  if (tid < HID) sh[tid] = hlast[b * HID + tid];
  __syncthreads();

  float logit[4];
#pragma unroll
  for (int j = 0; j < 4; ++j) {
    int o = tid + j * 256;
    float s = -1e30f;
    if (o < OUTSZ) {
      s = bd[o];
      const float* w = wd + (size_t)o * HID;
#pragma unroll 4
      for (int k = 0; k < HID; ++k) s = __fmaf_rn(sh[k], w[k], s);
    }
    logit[j] = s;
  }

  float m = fmaxf(fmaxf(logit[0], logit[1]), fmaxf(logit[2], logit[3]));
  red[tid] = m; __syncthreads();
  for (int s = 128; s > 0; s >>= 1) {
    if (tid < s) red[tid] = fmaxf(red[tid], red[tid + s]);
    __syncthreads();
  }
  m = red[0]; __syncthreads();

  float e[4], sum = 0.0f;
#pragma unroll
  for (int j = 0; j < 4; ++j) {
    int o = tid + j * 256;
    e[j] = (o < OUTSZ) ? __expf(logit[j] - m) : 0.0f;
    sum += e[j];
  }
  red[tid] = sum; __syncthreads();
  for (int s = 128; s > 0; s >>= 1) {
    if (tid < s) red[tid] += red[tid + s];
    __syncthreads();
  }
  float inv = 1.0f / red[0];
#pragma unroll
  for (int j = 0; j < 4; ++j) {
    int o = tid + j * 256;
    if (o < OUTSZ) out[(size_t)b * OUTSZ + o] = e[j] * inv;
  }
}

extern "C" void kernel_launch(void* const* d_in, const int* in_sizes, int n_in,
                              void* d_out, int out_size, void* d_ws, size_t ws_size,
                              hipStream_t stream) {
  const float* x   = (const float*)d_in[0];
  const float* wih = (const float*)d_in[1];
  const float* whh = (const float*)d_in[2];
  const float* bih = (const float*)d_in[3];
  const float* bhh = (const float*)d_in[4];
  const float* wd  = (const float*)d_in[5];
  const float* bd  = (const float*)d_in[6];
  float* out = (float*)d_out;

  // Workspace layout (~256.5 MB total)
  char* ws = (char*)d_ws;
  float* xproj = (float*)ws;                         // 512*256*512*4 = 268435456 B
  bf16*  wihb  = (bf16*)(ws + 268435456);            // 262144 B
  bf16*  whhb  = (bf16*)(ws + 268697600);            // 131072 B
  float* biasc = (float*)(ws + 268828672);           // 2048 B
  float* hlast = (float*)(ws + 268830720);           // 131072 B

  k_convert<<<512, 256, 0, stream>>>(wih, whh, bih, bhh, wihb, whhb, biasc);
  k_xproj<<<2048, 512, 0, stream>>>(x, wihb, biasc, xproj);
  k_lstm<<<16, 256, 0, stream>>>(xproj, whhb, hlast);
  k_dense<<<256, 256, 0, stream>>>(hlast, wd, bd, out);
}
